// SimpleTransformerBlock_37898791420010
// MI455X (gfx1250) — matrix operand order repaired and checked by reading, not compiled, not run on hardware
//
#include <hip/hip_runtime.h>
#include <hip/hip_bf16.h>

// ---------------------------------------------------------------------------
// SimpleTransformerBlock for MI455X (gfx1250, wave32, WMMA).
//
// Precision: split-bf16 "3-GEMM" emulation of fp32 GEMMs:
//   A*B ~= Ahi*Bhi + Ahi*Blo + Alo*Bhi   (f32 WMMA accumulators)
// using v_wmma_f32_16x16x32_bf16 (16384 FLOP/instr, 8x the per-instr FLOPs of
// the native f32 16x16x4 WMMA path). Residual error ~2^-18 => fp32-class.
//
// Tiling: block 128x256, 8 wave32s in a 2x4 grid of 64x64 wave tiles.
// Each wave: 4x4 WMMA subtiles -> 48 WMMA per K-step per wave against 16KB of
// LDS fragment reads = 48 FLOP/LDS-byte, matched to ~256 B/clk LDS vs the
// XDL WMMA issue rate (vs only 24 FLOP/byte for a 32x32 wave tile).
// ---------------------------------------------------------------------------

typedef __bf16 bf16_t;
typedef __attribute__((ext_vector_type(16))) __bf16 v16bf;
typedef __attribute__((ext_vector_type(8)))  float  v8f;

#define H_DIM  2048
#define B_DIM  2
#define S_DIM  2048
#define R_DIM  (B_DIM * S_DIM)   // 4096 token rows
#define H3_DIM (3 * H_DIM)       // 6144
#define H4_DIM (4 * H_DIM)       // 8192

constexpr int BMt  = 128;        // block tile M
constexpr int BNt  = 256;        // block tile N
constexpr int BKt  = 32;         // K-step
constexpr int LDAS = BKt + 4;    // padded LDS strides (bank spread)
constexpr int LDBS = BKt + 4;

enum { EPI_F32 = 0, EPI_SPLIT = 1, EPI_RES = 2, EPI_GELU_SPLIT = 3 };

union FragBF { v16bf v; unsigned u[8]; };

__device__ inline float gelu_exact(float x) {
    return 0.5f * x * (1.0f + erff(x * 0.70710678118654752f));
}

__device__ inline float block_sum(float v, float* red) {
    int t = threadIdx.x;
    red[t] = v; __syncthreads();
    for (int off = 128; off > 0; off >>= 1) {
        if (t < off) red[t] += red[t + off];
        __syncthreads();
    }
    float r = red[0]; __syncthreads();
    return r;
}

__device__ inline float block_max(float v, float* red) {
    int t = threadIdx.x;
    red[t] = v; __syncthreads();
    for (int off = 128; off > 0; off >>= 1) {
        if (t < off) red[t] = fmaxf(red[t], red[t + off]);
        __syncthreads();
    }
    float r = red[0]; __syncthreads();
    return r;
}

// ---------------------------------------------------------------------------
// fp32 -> (bf16 hi, bf16 lo) elementwise split (weights, once per launch)
// ---------------------------------------------------------------------------
__global__ __launch_bounds__(256)
void split_f32_kernel(const float* __restrict__ w,
                      bf16_t* __restrict__ hi, bf16_t* __restrict__ lo, size_t n) {
    size_t stride = (size_t)gridDim.x * blockDim.x;
    for (size_t i = (size_t)blockIdx.x * blockDim.x + threadIdx.x; i < n; i += stride) {
        float v = w[i];
        bf16_t h = (bf16_t)v;
        hi[i] = h;
        lo[i] = (bf16_t)(v - (float)h);
    }
}

// ---------------------------------------------------------------------------
// Row LayerNorm (cols == 2048) -> split bf16 outputs
// ---------------------------------------------------------------------------
__global__ __launch_bounds__(256)
void layernorm_split_kernel(const float* __restrict__ x,
                            const float* __restrict__ g, const float* __restrict__ b,
                            bf16_t* __restrict__ oh, bf16_t* __restrict__ ol, int cols) {
    __shared__ float red[256];
    int row = blockIdx.x;
    const float* xr = x + (size_t)row * cols;
    float v[8];
    float s = 0.f;
#pragma unroll
    for (int i = 0; i < 8; ++i) { v[i] = xr[threadIdx.x + 256 * i]; s += v[i]; }
    float mu = block_sum(s, red) * (1.0f / (float)H_DIM);
    float var = 0.f;
#pragma unroll
    for (int i = 0; i < 8; ++i) { float d = v[i] - mu; var += d * d; }
    var = block_sum(var, red) * (1.0f / (float)H_DIM);
    float rstd = rsqrtf(var + 1e-5f);
#pragma unroll
    for (int i = 0; i < 8; ++i) {
        int c = threadIdx.x + 256 * i;
        float y = (v[i] - mu) * rstd * g[c] + b[c];
        bf16_t h = (bf16_t)y;
        size_t idx = (size_t)row * cols + c;
        oh[idx] = h;
        ol[idx] = (bf16_t)(y - (float)h);
    }
}

// ---------------------------------------------------------------------------
// Row softmax (cols == 2048) -> split bf16 probabilities
// ---------------------------------------------------------------------------
__global__ __launch_bounds__(256)
void softmax_split_kernel(const float* __restrict__ sc,
                          bf16_t* __restrict__ oh, bf16_t* __restrict__ ol, int cols) {
    __shared__ float red[256];
    int row = blockIdx.x;
    const float* sr = sc + (size_t)row * cols;
    float v[8];
    float mx = -3.4e38f;
#pragma unroll
    for (int i = 0; i < 8; ++i) { v[i] = sr[threadIdx.x + 256 * i]; mx = fmaxf(mx, v[i]); }
    mx = block_max(mx, red);
    float sum = 0.f;
#pragma unroll
    for (int i = 0; i < 8; ++i) { v[i] = __expf(v[i] - mx); sum += v[i]; }
    sum = block_sum(sum, red);
    float inv = 1.0f / sum;
#pragma unroll
    for (int i = 0; i < 8; ++i) {
        int c = threadIdx.x + 256 * i;
        float p = v[i] * inv;
        bf16_t h = (bf16_t)p;
        size_t idx = (size_t)row * cols + c;
        oh[idx] = h;
        ol[idx] = (bf16_t)(p - (float)h);
    }
}

// ---------------------------------------------------------------------------
// Split-bf16 WMMA GEMM.
//   C[M,N] = scale * (A[M,K] @ op(B)) (+bias) (+epilogue)
//   BT=false: B is [K,N] row-major.   BT=true: B is [N,K] row-major (NT GEMM).
// Requires M%128==0, N%256==0, K%32==0 (true for all shapes here).
// ---------------------------------------------------------------------------
template <int EPI, bool BT>
__global__ __launch_bounds__(256)
void gemm_split_wmma(int M, int N, int K,
                     const bf16_t* __restrict__ Ah, const bf16_t* __restrict__ Al, int lda,
                     const bf16_t* __restrict__ Bh, const bf16_t* __restrict__ Bl, int ldb,
                     const float* __restrict__ bias, float scale,
                     const float* __restrict__ res,
                     float* __restrict__ Cf,
                     bf16_t* __restrict__ Chi, bf16_t* __restrict__ Clo, int ldc) {
    (void)M; (void)N;
    __shared__ bf16_t AsH[BMt * LDAS], AsL[BMt * LDAS];   // 9.2 KB each
    __shared__ bf16_t BsH[BNt * LDBS], BsL[BNt * LDBS];   // 18.4 KB each

    const int tid  = threadIdx.x;
    const int lane = tid & 31;
    const int wave = tid >> 5;
    const int m0 = blockIdx.y * BMt;
    const int n0 = blockIdx.x * BNt;
    const int wm = (wave & 1) * 64;   // 2 wave rows
    const int wn = (wave >> 1) * 64;  // 4 wave cols
    const int l16  = lane & 15;
    const int half = lane >> 4;

    v8f acc[4][4] = {};   // 64x64 wave tile = 128 accumulator VGPRs

    for (int k0 = 0; k0 < K; k0 += BKt) {
        // ---- stage A tile (128 x 32, hi+lo), 16B vectors along K ----
#pragma unroll
        for (int it = 0; it < 2; ++it) {
            int s   = tid + it * 256;          // 512 slots
            int row = s >> 2;
            int kc  = (s & 3) * 8;
            size_t go = (size_t)(m0 + row) * lda + k0 + kc;
            uint4 vh = *(const uint4*)(Ah + go);
            uint4 vl = *(const uint4*)(Al + go);
            unsigned* dh = (unsigned*)&AsH[row * LDAS + kc];
            unsigned* dl = (unsigned*)&AsL[row * LDAS + kc];
            dh[0] = vh.x; dh[1] = vh.y; dh[2] = vh.z; dh[3] = vh.w;
            dl[0] = vl.x; dl[1] = vl.y; dl[2] = vl.z; dl[3] = vl.w;
            if (k0 + BKt < K) __builtin_prefetch(Ah + go + BKt, 0, 1);  // global_prefetch_b8
        }
        // ---- stage B tile (32 x 256 logical) into LDS as Bs[n][k] ----
        if constexpr (!BT) {
            // Global B is [K,N]: read 8 contiguous n, scatter-transpose into LDS.
#pragma unroll
            for (int it = 0; it < 4; ++it) {
                int s  = tid + it * 256;       // 1024 slots
                int kk = s >> 5;               // 32 k rows
                int nc = (s & 31) * 8;         // 32 chunks of 8 cols
                size_t go = (size_t)(k0 + kk) * ldb + n0 + nc;
                uint4 vh = *(const uint4*)(Bh + go);
                uint4 vl = *(const uint4*)(Bl + go);
                const bf16_t* eh = (const bf16_t*)&vh;
                const bf16_t* el = (const bf16_t*)&vl;
#pragma unroll
                for (int j = 0; j < 8; ++j) {
                    BsH[(nc + j) * LDBS + kk] = eh[j];
                    BsL[(nc + j) * LDBS + kk] = el[j];
                }
                if (k0 + BKt < K) __builtin_prefetch(Bh + go + (size_t)BKt * ldb, 0, 1);
            }
        } else {
            // Global B is [N,K]: straight copy, vectorized along K.
#pragma unroll
            for (int it = 0; it < 4; ++it) {
                int s   = tid + it * 256;      // 1024 slots
                int row = s >> 2;              // 256 n rows
                int kc  = (s & 3) * 8;
                size_t go = (size_t)(n0 + row) * ldb + k0 + kc;
                uint4 vh = *(const uint4*)(Bh + go);
                uint4 vl = *(const uint4*)(Bl + go);
                unsigned* dh = (unsigned*)&BsH[row * LDBS + kc];
                unsigned* dl = (unsigned*)&BsL[row * LDBS + kc];
                dh[0] = vh.x; dh[1] = vh.y; dh[2] = vh.z; dh[3] = vh.w;
                dl[0] = vl.x; dl[1] = vl.y; dl[2] = vl.z; dl[3] = vl.w;
                if (k0 + BKt < K) __builtin_prefetch(Bh + go + BKt, 0, 1);
            }
        }
        __syncthreads();

        // ---- A fragments for all 4 M-subtiles (reused across 4 N-subtiles) ----
        // A 16x32 layout: lane m=l%16; VGPR v holds K pair k=(v/4)*16+(l/16)*8+(v%4)*2
        FragBF fah[4], fal[4];
#pragma unroll
        for (int mt = 0; mt < 4; ++mt) {
            const bf16_t* ph = &AsH[(wm + mt * 16 + l16) * LDAS];
            const bf16_t* pl = &AsL[(wm + mt * 16 + l16) * LDAS];
#pragma unroll
            for (int v = 0; v < 8; ++v) {
                int kk = ((v >> 2) << 4) + (half << 3) + ((v & 3) << 1);
                fah[mt].u[v] = *(const unsigned*)(ph + kk);
                fal[mt].u[v] = *(const unsigned*)(pl + kk);
            }
        }
        // ---- per N-subtile: load B frags once, feed 4 M-subtiles ----
        // B 32x16 layout: lane n=l%16; VGPR v holds K pair k=(v/4)*16+(l/16)*8+(v%4)*2
#pragma unroll
        for (int nt = 0; nt < 4; ++nt) {
            FragBF fbh, fbl;
            const bf16_t* ph = &BsH[(wn + nt * 16 + l16) * LDBS];
            const bf16_t* pl = &BsL[(wn + nt * 16 + l16) * LDBS];
#pragma unroll
            for (int v = 0; v < 8; ++v) {
                int kk = ((v >> 2) << 4) + (half << 3) + ((v & 3) << 1);
                fbh.u[v] = *(const unsigned*)(ph + kk);
                fbl.u[v] = *(const unsigned*)(pl + kk);
            }
#pragma unroll
            for (int mt = 0; mt < 4; ++mt) {
                acc[mt][nt] = __builtin_amdgcn_wmma_f32_16x16x32_bf16(
                    false, fal[mt].v, false, fbh.v, (short)0, acc[mt][nt], false, false);
                acc[mt][nt] = __builtin_amdgcn_wmma_f32_16x16x32_bf16(
                    false, fah[mt].v, false, fbl.v, (short)0, acc[mt][nt], false, false);
                acc[mt][nt] = __builtin_amdgcn_wmma_f32_16x16x32_bf16(
                    false, fah[mt].v, false, fbh.v, (short)0, acc[mt][nt], false, false);
            }
        }
        __syncthreads();
    }

    // ---- epilogue (C layout: VGPR r, lane l -> row r + 8*(l/16), col l%16) ----
    const int rbase = half * 8;
#pragma unroll
    for (int mt = 0; mt < 4; ++mt)
#pragma unroll
        for (int nt = 0; nt < 4; ++nt) {
            int gn = n0 + wn + nt * 16 + l16;
#pragma unroll
            for (int r = 0; r < 8; ++r) {
                int gm = m0 + wm + mt * 16 + rbase + r;
                float c = acc[mt][nt][r] * scale;
                if (bias) c += bias[gn];
                size_t idx = (size_t)gm * ldc + gn;
                if constexpr (EPI == EPI_F32) {
                    Cf[idx] = c;
                } else if constexpr (EPI == EPI_RES) {
                    Cf[idx] = c + res[idx];
                } else {
                    if constexpr (EPI == EPI_GELU_SPLIT) c = gelu_exact(c);
                    bf16_t h = (bf16_t)c;
                    Chi[idx] = h;
                    Clo[idx] = (bf16_t)(c - (float)h);
                }
            }
        }
}

// ---------------------------------------------------------------------------
// Host-side orchestration
// ---------------------------------------------------------------------------
extern "C" void kernel_launch(void* const* d_in, const int* in_sizes, int n_in,
                              void* d_out, int out_size, void* d_ws, size_t ws_size,
                              hipStream_t stream) {
    (void)in_sizes; (void)n_in; (void)out_size; (void)ws_size;

    const float* x      = (const float*)d_in[0];
    const float* ln1_g  = (const float*)d_in[1];
    const float* ln1_b  = (const float*)d_in[2];
    const float* ln2_g  = (const float*)d_in[3];
    const float* ln2_b  = (const float*)d_in[4];
    const float* qkv_w  = (const float*)d_in[5];
    const float* qkv_b  = (const float*)d_in[6];
    const float* out_w  = (const float*)d_in[7];
    const float* out_b  = (const float*)d_in[8];
    const float* mlp_w1 = (const float*)d_in[9];
    const float* mlp_b1 = (const float*)d_in[10];
    const float* mlp_w2 = (const float*)d_in[11];
    const float* mlp_b2 = (const float*)d_in[12];

    // Workspace layout (all sizes multiples of 256B). Total ~503 MB.
    char* base = (char*)d_ws;
    size_t cur = 0;
    auto take = [&](size_t bytes) { char* p = base + cur; cur += bytes; return p; };

    bf16_t* wqkv_h = (bf16_t*)take((size_t)H_DIM * H3_DIM * 2);
    bf16_t* wqkv_l = (bf16_t*)take((size_t)H_DIM * H3_DIM * 2);
    bf16_t* wout_h = (bf16_t*)take((size_t)H_DIM * H_DIM * 2);
    bf16_t* wout_l = (bf16_t*)take((size_t)H_DIM * H_DIM * 2);
    bf16_t* w1_h   = (bf16_t*)take((size_t)H_DIM * H4_DIM * 2);
    bf16_t* w1_l   = (bf16_t*)take((size_t)H_DIM * H4_DIM * 2);
    bf16_t* w2_h   = (bf16_t*)take((size_t)H4_DIM * H_DIM * 2);
    bf16_t* w2_l   = (bf16_t*)take((size_t)H4_DIM * H_DIM * 2);
    bf16_t* h1_h   = (bf16_t*)take((size_t)R_DIM * H_DIM * 2);
    bf16_t* h1_l   = (bf16_t*)take((size_t)R_DIM * H_DIM * 2);
    bf16_t* qkv_h  = (bf16_t*)take((size_t)R_DIM * H3_DIM * 2);
    bf16_t* qkv_l  = (bf16_t*)take((size_t)R_DIM * H3_DIM * 2);
    float*  scores = (float*) take((size_t)B_DIM * S_DIM * S_DIM * 4);
    bf16_t* p_h    = (bf16_t*)take((size_t)B_DIM * S_DIM * S_DIM * 2);
    bf16_t* p_l    = (bf16_t*)take((size_t)B_DIM * S_DIM * S_DIM * 2);
    bf16_t* ao_h   = (bf16_t*)take((size_t)R_DIM * H_DIM * 2);
    bf16_t* ao_l   = (bf16_t*)take((size_t)R_DIM * H_DIM * 2);
    float*  y1     = (float*) take((size_t)R_DIM * H_DIM * 4);
    bf16_t* h2_h   = (bf16_t*)take((size_t)R_DIM * H_DIM * 2);
    bf16_t* h2_l   = (bf16_t*)take((size_t)R_DIM * H_DIM * 2);
    // MLP hidden (134.2 MB) aliases the dead qkv+scores region (exactly 134.2 MB)
    bf16_t* g_h = qkv_h;
    bf16_t* g_l = qkv_h + (size_t)R_DIM * H4_DIM;

    const dim3 blk(256);

    // 1) weights fp32 -> bf16 hi/lo
    split_f32_kernel<<<4096, blk, 0, stream>>>(qkv_w,  wqkv_h, wqkv_l, (size_t)H_DIM * H3_DIM);
    split_f32_kernel<<<2048, blk, 0, stream>>>(out_w,  wout_h, wout_l, (size_t)H_DIM * H_DIM);
    split_f32_kernel<<<4096, blk, 0, stream>>>(mlp_w1, w1_h,   w1_l,   (size_t)H_DIM * H4_DIM);
    split_f32_kernel<<<4096, blk, 0, stream>>>(mlp_w2, w2_h,   w2_l,   (size_t)H4_DIM * H_DIM);

    // 2) LN1
    layernorm_split_kernel<<<R_DIM, blk, 0, stream>>>(x, ln1_g, ln1_b, h1_h, h1_l, H_DIM);

    // 3) QKV projection: [4096,2048] @ [2048,6144] + b -> split bf16
    gemm_split_wmma<EPI_SPLIT, false><<<dim3(H3_DIM / BNt, R_DIM / BMt), blk, 0, stream>>>(
        R_DIM, H3_DIM, H_DIM, h1_h, h1_l, H_DIM, wqkv_h, wqkv_l, H3_DIM,
        qkv_b, 1.0f, nullptr, nullptr, qkv_h, qkv_l, H3_DIM);

    // 4) scores = q @ k^T / sqrt(H)  (NT GEMM, per batch)
    const float iscale = 0.022097086912079608f;  // 1/sqrt(2048)
    for (int b = 0; b < B_DIM; ++b) {
        size_t qo = (size_t)b * S_DIM * H3_DIM;
        gemm_split_wmma<EPI_F32, true><<<dim3(S_DIM / BNt, S_DIM / BMt), blk, 0, stream>>>(
            S_DIM, S_DIM, H_DIM,
            qkv_h + qo, qkv_l + qo, H3_DIM,                    // q
            qkv_h + qo + H_DIM, qkv_l + qo + H_DIM, H3_DIM,    // k (row-major [S,H])
            nullptr, iscale, nullptr, scores + (size_t)b * S_DIM * S_DIM,
            nullptr, nullptr, S_DIM);
    }

    // 5) softmax rows -> split bf16 probabilities
    softmax_split_kernel<<<B_DIM * S_DIM, blk, 0, stream>>>(scores, p_h, p_l, S_DIM);

    // 6) attn_out = p @ v (NN GEMM, per batch)
    for (int b = 0; b < B_DIM; ++b) {
        size_t po = (size_t)b * S_DIM * S_DIM;
        size_t vo = (size_t)b * S_DIM * H3_DIM + 2 * H_DIM;
        size_t oo = (size_t)b * S_DIM * H_DIM;
        gemm_split_wmma<EPI_SPLIT, false><<<dim3(H_DIM / BNt, S_DIM / BMt), blk, 0, stream>>>(
            S_DIM, H_DIM, S_DIM, p_h + po, p_l + po, S_DIM,
            qkv_h + vo, qkv_l + vo, H3_DIM,
            nullptr, 1.0f, nullptr, nullptr, ao_h + oo, ao_l + oo, H_DIM);
    }

    // 7) y1 = attn_out @ out_w + out_b + x   (residual 1)
    gemm_split_wmma<EPI_RES, false><<<dim3(H_DIM / BNt, R_DIM / BMt), blk, 0, stream>>>(
        R_DIM, H_DIM, H_DIM, ao_h, ao_l, H_DIM, wout_h, wout_l, H_DIM,
        out_b, 1.0f, x, y1, nullptr, nullptr, H_DIM);

    // 8) LN2
    layernorm_split_kernel<<<R_DIM, blk, 0, stream>>>(y1, ln2_g, ln2_b, h2_h, h2_l, H_DIM);

    // 9) g = gelu(h2 @ mlp_w1 + b1) -> split bf16 (aliases dead qkv/scores)
    gemm_split_wmma<EPI_GELU_SPLIT, false><<<dim3(H4_DIM / BNt, R_DIM / BMt), blk, 0, stream>>>(
        R_DIM, H4_DIM, H_DIM, h2_h, h2_l, H_DIM, w1_h, w1_l, H4_DIM,
        mlp_b1, 1.0f, nullptr, nullptr, g_h, g_l, H4_DIM);

    // 10) out = g @ mlp_w2 + b2 + y1   (residual 2) -> d_out fp32
    gemm_split_wmma<EPI_RES, false><<<dim3(H_DIM / BNt, R_DIM / BMt), blk, 0, stream>>>(
        R_DIM, H_DIM, H4_DIM, g_h, g_l, H4_DIM, w2_h, w2_l, H_DIM,
        mlp_b2, 1.0f, y1, (float*)d_out, nullptr, nullptr, H_DIM);
}
